// MKCapture_3616362463456
// MI455X (gfx1250) — matrix operand order repaired and checked by reading, not compiled, not run on hardware
//
#include <hip/hip_runtime.h>

// ---------------- problem constants (from reference) ----------------
#define BATCH   8192
#define NSTEP   64
#define HID     64
#define PDIM    128
#define NUMR    11
#define GAP     10
#define RMIN_I  10
#define MC      10000
#define DT_F    (1.0f/64.0f)
#define RDOM_F  5.0f
#define CAPEPS  0.1f
#define PI_F    3.14159265358979323846f
// sqrt(2*DXC)=sqrt(2*DYC)=1.0 exactly (DXC=DYC=0.5)

// packed per-step weight block (f16, WMMA-B-fragment order), in f16 elements
#define SEG_G1  0       // W1  padded to 32x64 : 4 nt * 32 lanes * 16 = 2048
#define SEG_G2  2048    // W2  64x64           : 2 kt * 4 nt * 512   = 4096
#define SEG_F1  6144    // Wf1 padded          : 2048
#define SEG_F2  8192    // Wf2                 : 4096
#define WBLK    12288   // f16 elements per step  (24 KB)

typedef __attribute__((ext_vector_type(16))) _Float16 v16h;
typedef __attribute__((ext_vector_type(8)))  float    v8f;
typedef unsigned int v4u __attribute__((ext_vector_type(4)));
typedef int          v8i __attribute__((ext_vector_type(8)));
typedef int          v4i __attribute__((ext_vector_type(4)));

union Frag { v16h v; _Float16 h[16]; unsigned u[8]; };

// branch-free tanh: one v_exp_f32 + one v_rcp_f32 + FMA (no IEEE div sequence)
__device__ __forceinline__ float tanh_fast(float x)
{
    x = fminf(15.0f, fmaxf(-15.0f, x));
    float e = __expf(2.0f * x);
    float r = __builtin_amdgcn_rcpf(e + 1.0f);
    return __builtin_fmaf(-2.0f, r, 1.0f);
}

// ---------------- r-chain precompute: ridx/drt per (t, path) ----------------
__global__ __launch_bounds__(256)
void k_rchain(const int* __restrict__ rt0,
              const float* __restrict__ jump_unif,
              const float* __restrict__ size_unif,
              const float* __restrict__ jm,       // [NUMR][2*GAP]
              const float* __restrict__ cr,       // [NUMR]
              signed char* __restrict__ ridx8,
              signed char* __restrict__ drt8)
{
    int b = blockIdx.x * blockDim.x + threadIdx.x;
    if (b >= BATCH) return;
    int rt = rt0[b];
    for (int t = 0; t < NSTEP; ++t) {
        int r = rt - RMIN_I;
        r = r < 0 ? 0 : (r > GAP ? GAP : r);
        float ju = jump_unif[t * BATCH + b];
        int drt = 0;
        if (ju < cr[r] * DT_F) {
            float su = size_unif[t * BATCH + b];
            int cnt = 0;
            #pragma unroll
            for (int j = 0; j < 2 * GAP; ++j) cnt += (su < jm[r * (2 * GAP) + j]) ? 1 : 0;
            int ind = 2 * GAP - cnt;
            drt = (ind < GAP) ? (ind + 1) : -(ind - GAP + 1);
        }
        ridx8[t * BATCH + b] = (signed char)r;
        drt8 [t * BATCH + b] = (signed char)drt;
        rt += drt;
        rt = rt < RMIN_I ? RMIN_I : (rt > RMIN_I + GAP ? RMIN_I + GAP : rt);
    }
}

// ---------------- mc_jump: histogram then weighted build ----------------
__global__ void k_zero_i32(int* p, int n)
{
    int i = blockIdx.x * blockDim.x + threadIdx.x;
    if (i < n) p[i] = 0;
}

__global__ __launch_bounds__(256)
void k_hist(const float* __restrict__ mc_unif,
            const float* __restrict__ jm,
            int* __restrict__ hist)               // [NUMR][2][GAP]
{
    int i = blockIdx.x * blockDim.x + threadIdx.x;
    if (i >= MC * NUMR) return;
    int r = i % NUMR;
    float u = mc_unif[i];
    int cnt = 0;
    #pragma unroll
    for (int j = 0; j < 2 * GAP; ++j) cnt += (u < jm[r * (2 * GAP) + j]) ? 1 : 0;
    int ind = 2 * GAP - cnt;
    int s = (ind < GAP) ? (ind + 1) : -(ind - GAP + 1);
    int sgn = (s > 0) ? 0 : 1;
    int a = (s > 0 ? s : -s) - 1;
    atomicAdd(&hist[(r * 2 + sgn) * GAP + a], 1);
}

__global__ __launch_bounds__(PDIM)
void k_mcbuild(const int* __restrict__ hist,
               const float* __restrict__ jump_l,  // [2][GAP][P]
               const float* __restrict__ cr,
               float* __restrict__ mc_jump)       // [NUMR][P]
{
    int r = blockIdx.x, p = threadIdx.x;
    float acc = 0.f;
    #pragma unroll
    for (int a = 0; a < GAP; ++a) {
        acc += (float)hist[(r * 2 + 0) * GAP + a] * jump_l[(0 * GAP + a) * PDIM + p];
        acc += (float)hist[(r * 2 + 1) * GAP + a] * jump_l[(1 * GAP + a) * PDIM + p];
    }
    mc_jump[r * PDIM + p] = acc * (1.0f / (float)MC) * cr[r];
}

// ---------------- fold emb into W3 ----------------
__global__ __launch_bounds__(PDIM)
void k_w3r(const float* __restrict__ emb, const float* __restrict__ W3,
           const float* __restrict__ b3,
           float* __restrict__ W3r, float* __restrict__ b3r)
{
    int t = blockIdx.x / NUMR, r = blockIdx.x % NUMR;
    int tid = threadIdx.x;
    __shared__ float e[PDIM];
    e[tid] = emb[(t * NUMR + r) * PDIM + tid];
    __syncthreads();
    for (int out = tid; out < HID * 6; out += PDIM) {
        int k = out / 6, o = out % 6;
        const float* w = W3 + (size_t)(t * HID + k) * (PDIM * 6) + o;
        float acc = 0.f;
        for (int p = 0; p < PDIM; ++p) acc += e[p] * w[p * 6];
        W3r[((t * NUMR + r) * HID + k) * 6 + o] = acc;
    }
    if (tid < 6) {
        const float* bb = b3 + (size_t)t * (PDIM * 6) + tid;
        float acc = 0.f;
        for (int p = 0; p < PDIM; ++p) acc += e[p] * bb[p * 6];
        b3r[(t * NUMR + r) * 6 + tid] = acc;
    }
}

// ---------------- fold jump weights ----------------
__global__ __launch_bounds__(HID)
void k_wf3r(const float* __restrict__ Wf3, const float* __restrict__ bf3,
            const float* __restrict__ jump_r, const float* __restrict__ jump_l,
            const float* __restrict__ mc_jump,
            float* __restrict__ Cjp, float* __restrict__ c0jp)
{
    int t   = blockIdx.x / (NUMR * 21);
    int rem = blockIdx.x % (NUMR * 21);
    int r   = rem / 21;
    int d   = rem % 21 - GAP;
    int tid = threadIdx.x;
    __shared__ float wv[PDIM];
    __shared__ float red[HID];
    for (int p = tid; p < PDIM; p += HID) {
        float jl = 0.f;
        if (d > 0)      jl = jump_l[(0 * GAP + (d - 1)) * PDIM + p];
        else if (d < 0) jl = jump_l[(1 * GAP + (-d - 1)) * PDIM + p];
        wv[p] = jump_r[r * PDIM + p] * (jl - mc_jump[r * PDIM + p] * DT_F);
    }
    __syncthreads();
    {
        const float* w = Wf3 + (size_t)(t * HID + tid) * PDIM;
        float acc = 0.f;
        for (int p = 0; p < PDIM; ++p) acc += w[p] * wv[p];
        Cjp[(size_t)blockIdx.x * HID + tid] = acc;
    }
    float s = 0.f;
    for (int p = tid; p < PDIM; p += HID) s += bf3[t * PDIM + p] * wv[p];
    red[tid] = s;
    __syncthreads();
    for (int off = HID / 2; off > 0; off >>= 1) {
        if (tid < off) red[tid] += red[tid + off];
        __syncthreads();
    }
    if (tid == 0) c0jp[blockIdx.x] = red[0];
}

// ---------------- pre-swizzle layer weights into WMMA B-fragment order (f16) ----------------
// B fragment (contiguous-k order used by this file, 16-bit B 32x16): lane l holds col = nt*16+(l&15);
// element e (0..15) holds B[k][col] with k = kt*32 + (l>>4)*16 + e.
__global__ __launch_bounds__(256)
void k_pack(const float* __restrict__ W1, const float* __restrict__ W2,
            const float* __restrict__ Wf1, const float* __restrict__ Wf2,
            _Float16* __restrict__ Wpack)
{
    int t = blockIdx.x;
    _Float16* dst = Wpack + (size_t)t * WBLK;
    for (int idx = threadIdx.x; idx < 2048; idx += 256) {          // G1 (kt=0)
        int nt = idx >> 9, lane = (idx >> 4) & 31, e = idx & 15;
        int col = nt * 16 + (lane & 15);
        int k = (lane >> 4) * 16 + e;
        dst[SEG_G1 + idx] = (k < 6) ? (_Float16)W1[(t * 6 + k) * HID + col] : (_Float16)0.f;
    }
    for (int idx = threadIdx.x; idx < 4096; idx += 256) {          // G2
        int kt = idx >> 11, nt = (idx >> 9) & 3, lane = (idx >> 4) & 31, e = idx & 15;
        int col = nt * 16 + (lane & 15);
        int k = kt * 32 + (lane >> 4) * 16 + e;
        dst[SEG_G2 + idx] = (_Float16)W2[(size_t)t * HID * HID + k * HID + col];
    }
    for (int idx = threadIdx.x; idx < 2048; idx += 256) {          // F1 (kt=0)
        int nt = idx >> 9, lane = (idx >> 4) & 31, e = idx & 15;
        int col = nt * 16 + (lane & 15);
        int k = (lane >> 4) * 16 + e;
        dst[SEG_F1 + idx] = (k < 6) ? (_Float16)Wf1[(t * 6 + k) * HID + col] : (_Float16)0.f;
    }
    for (int idx = threadIdx.x; idx < 4096; idx += 256) {          // F2
        int kt = idx >> 11, nt = (idx >> 9) & 3, lane = (idx >> 4) & 31, e = idx & 15;
        int col = nt * 16 + (lane & 15);
        int k = kt * 32 + (lane >> 4) * 16 + e;
        dst[SEG_F2 + idx] = (_Float16)Wf2[(size_t)t * HID * HID + k * HID + col];
    }
}

// ---------------- TDM: DMA one step's packed weight block (24 KB) into LDS ----------------
__device__ __forceinline__ void tdm_load_weights(unsigned lds_off, unsigned long long gaddr)
{
    const int dim0 = WBLK / 4;  // 3072 elements of data_size 8B
    v4u g0;
    g0[0] = 1u;                                                    // count=1 (valid descriptor)
    g0[1] = lds_off;                                               // lds_addr (bytes)
    g0[2] = (unsigned)(gaddr & 0xFFFFFFFFull);                     // global_addr[31:0]
    g0[3] = (unsigned)((gaddr >> 32) & 0x1FFFFFFull) | (2u << 30); // global_addr[56:32] | type=2
    v8i g1;
    g1[0] = (int)(3u << 16);                 // data_size=3 (8 bytes); no multicast/iterate/pad
    g1[1] = (int)((dim0 & 0xFFFF) << 16);    // tensor_dim0[15:0]
    g1[2] = (int)((dim0 >> 16) | (1 << 16)); // tensor_dim0[31:16] | tensor_dim1=1 (lo16)
    g1[3] = (int)(dim0 << 16);               // tensor_dim1 hi16=0 | tile_dim0=3072
    g1[4] = 0;                               // tile_dim1=0 (unused), tile_dim2=0
    g1[5] = dim0;                            // tensor_dim0_stride[31:0]
    g1[6] = 0;
    g1[7] = 0;
    v4i gz = {0, 0, 0, 0};
#if defined(__clang_major__) && (__clang_major__ >= 23)
    v8i gz8 = {0, 0, 0, 0, 0, 0, 0, 0};
    __builtin_amdgcn_tensor_load_to_lds(g0, g1, gz, gz, gz8, 0);
#else
    __builtin_amdgcn_tensor_load_to_lds(g0, g1, gz, gz, 0);
#endif
}

// ---------------- WMMA block GEMMs ----------------
// Layer-1: A = sIn (128 x 8 f16, K padded with zeros in registers), B fragments from LDS.
__device__ __forceinline__
void gemm1(const _Float16* sIn8, const _Float16* Bseg, const float* __restrict__ bias,
           _Float16* Out, int l16, int half, int wv, int lane)
{
    unsigned amask = half ? 0u : 0xFFFFFFFFu;   // lanes 16-31 cover K=16..31 -> zero, branchless
    for (int mt = wv * 2; mt < wv * 2 + 2; ++mt) {
        int row = mt * 16 + l16;
        const unsigned* ap = (const unsigned*)(sIn8 + row * 8);
        unsigned a0 = ap[0] & amask, a1 = ap[1] & amask, a2 = ap[2] & amask, a3 = ap[3] & amask;
        #pragma unroll
        for (int nt = 0; nt < 4; ++nt) {
            Frag a, b;
            a.u[0] = a0; a.u[1] = a1; a.u[2] = a2; a.u[3] = a3;
            #pragma unroll
            for (int v = 4; v < 8; ++v) a.u[v] = 0u;                 // K=8..15 / 24..31 (elements 8..15) -> 0
            const unsigned* bp = (const unsigned*)(Bseg + (size_t)(nt * 32 + lane) * 16);
            #pragma unroll
            for (int v = 0; v < 8; ++v) b.u[v] = bp[v];
            v8f acc = {};
            acc = __builtin_amdgcn_wmma_f32_16x16x32_f16(
                    false, a.v, false, b.v, (short)0, acc, false, false);
            int col = nt * 16 + l16;
            float bs = bias[col];
            #pragma unroll
            for (int v = 0; v < 8; ++v)
                Out[(mt * 16 + v + 8 * half) * HID + col] = (_Float16)tanh_fast(acc[v] + bs);
        }
    }
}

// Layer-2: A = 128 x 64 f16 activations in LDS, K = 2 k-tiles.
__device__ __forceinline__
void gemm2(const _Float16* A, const _Float16* Bseg, const float* __restrict__ bias,
           _Float16* Out, int l16, int half, int wv, int lane)
{
    for (int mt = wv * 2; mt < wv * 2 + 2; ++mt) {
        int row = mt * 16 + l16;
        #pragma unroll
        for (int nt = 0; nt < 4; ++nt) {
            v8f acc = {};
            #pragma unroll
            for (int kt = 0; kt < 2; ++kt) {
                Frag a, b;
                const unsigned* ap = (const unsigned*)(A + row * HID + kt * 32 + half * 16);
                #pragma unroll
                for (int v = 0; v < 4; ++v) a.u[v] = ap[v];
                #pragma unroll
                for (int v = 0; v < 4; ++v) a.u[4 + v] = ap[4 + v];
                const unsigned* bp = (const unsigned*)(Bseg + (size_t)((kt * 4 + nt) * 32 + lane) * 16);
                #pragma unroll
                for (int v = 0; v < 8; ++v) b.u[v] = bp[v];
                acc = __builtin_amdgcn_wmma_f32_16x16x32_f16(
                        false, a.v, false, b.v, (short)0, acc, false, false);
            }
            int col = nt * 16 + l16;
            float bs = bias[col];
            #pragma unroll
            for (int v = 0; v < 8; ++v)
                Out[(mt * 16 + v + 8 * half) * HID + col] = (_Float16)tanh_fast(acc[v] + bs);
        }
    }
}

// ---------------- reflection at the ball boundary ----------------
__device__ __forceinline__ void reflect3(float* pos, float* g)
{
    float nr = sqrtf(pos[0]*pos[0] + pos[1]*pos[1] + pos[2]*pos[2]);
    float inv = 1.0f / fmaxf(nr, 1e-12f);
    float nb0 = pos[0]*inv, nb1 = pos[1]*inv, nb2 = pos[2]*inv;
    if (nr > RDOM_F) {
        float sc = 2.0f * RDOM_F - nr;
        pos[0] = nb0 * sc; pos[1] = nb1 * sc; pos[2] = nb2 * sc;
        float pj0 = nb0*g[0] + nb1*g[3] + nb2*g[6];
        float pj1 = nb0*g[1] + nb1*g[4] + nb2*g[7];
        float pj2 = nb0*g[2] + nb1*g[5] + nb2*g[8];
        g[0] -= 2.f*nb0*pj0; g[1] -= 2.f*nb0*pj1; g[2] -= 2.f*nb0*pj2;
        g[3] -= 2.f*nb1*pj0; g[4] -= 2.f*nb1*pj1; g[5] -= 2.f*nb1*pj2;
        g[6] -= 2.f*nb2*pj0; g[7] -= 2.f*nb2*pj1; g[8] -= 2.f*nb2*pj2;
    }
}

// ---------------- main persistent scan ----------------
__global__ __launch_bounds__(128)
void k_main(const signed char* __restrict__ ridx8,
            const signed char* __restrict__ drt8,
            const _Float16* __restrict__ Wpack,
            const float* __restrict__ b1,  const float* __restrict__ b2,
            const float* __restrict__ bf1, const float* __restrict__ bf2,
            const float* __restrict__ W3r, const float* __restrict__ b3r,
            const float* __restrict__ Cjp, const float* __restrict__ c0jp,
            const float* __restrict__ uin,
            const float* __restrict__ xt0, const float* __restrict__ yt0,
            const float* __restrict__ dBxt, const float* __restrict__ dByt,
            const float* __restrict__ cfr,
            float* __restrict__ out)
{
    __shared__ _Float16 sWgt[WBLK];      // 24 KB: this step's weights, B-fragment order (TDM fed)
    __shared__ _Float16 sIn [128 * 8];   //  2 KB: inp6 padded to 8
    __shared__ _Float16 sH1 [128 * HID]; // 16 KB
    __shared__ _Float16 sH2 [128 * HID]; // 16 KB

    const int tid  = threadIdx.x;
    const int lane = tid & 31;
    const int l16  = lane & 15;
    const int half = lane >> 4;
    const int wv   = __builtin_amdgcn_readfirstlane((int)(threadIdx.x >> 5)); // scalar wave id
    const int gid  = blockIdx.x * 128 + tid;

    const unsigned lds_wgt = (unsigned)(unsigned long long)(uintptr_t)sWgt; // LDS byte offset

    float xt[3], xin[3], yin[3], gtx[9], gty[9];
    #pragma unroll
    for (int i = 0; i < 3; ++i) {
        xt[i]  = xt0[gid * 3 + i];
        xin[i] = xt[i];
        yin[i] = yt0[gid * 3 + i];
    }
    #pragma unroll
    for (int i = 0; i < 9; ++i) { gtx[i] = (i % 4 == 0) ? 1.f : 0.f; gty[i] = gtx[i]; }
    float u_pre = uin[0], fun = 0.f, run = 1.f;

    sIn[tid * 8 + 6] = (_Float16)0.f;    // pad columns, written once
    sIn[tid * 8 + 7] = (_Float16)0.f;

    for (int t = 0; t < NSTEP; ++t) {
        int r = ridx8[t * BATCH + gid];
        int d = drt8 [t * BATCH + gid];

        // async weight DMA for this step (wave 0 issues; everyone syncs on the result)
        if (wv == 0)
            tdm_load_weights(lds_wgt, (unsigned long long)(uintptr_t)(Wpack + (size_t)t * WBLK));

        sIn[tid * 8 + 0] = (_Float16)xin[0];
        sIn[tid * 8 + 1] = (_Float16)xin[1];
        sIn[tid * 8 + 2] = (_Float16)xin[2];
        sIn[tid * 8 + 3] = (_Float16)yin[0];
        sIn[tid * 8 + 4] = (_Float16)yin[1];
        sIn[tid * 8 + 5] = (_Float16)yin[2];

        __builtin_amdgcn_s_wait_tensorcnt((short)0);
        __syncthreads();

        // ---- g-MLP ----
        gemm1(sIn, sWgt + SEG_G1, b1 + t * HID, sH1, l16, half, wv, lane);
        __syncthreads();
        gemm2(sH1, sWgt + SEG_G2, b2 + t * HID, sH2, l16, half, wv, lane);
        __syncthreads();

        // exit: gu = h2 . W3r[t][r] + b3r[t][r]
        float gu[6] = {0.f, 0.f, 0.f, 0.f, 0.f, 0.f};
        {
            const float* w3p = W3r + (size_t)((t * NUMR + r) * HID) * 6;
            for (int k = 0; k < HID; ++k) {
                float h = (float)sH2[tid * HID + k];
                #pragma unroll
                for (int o = 0; o < 6; ++o) gu[o] += h * w3p[k * 6 + o];
            }
            const float* b3p = b3r + (t * NUMR + r) * 6;
            #pragma unroll
            for (int o = 0; o < 6; ++o) gu[o] += b3p[o];
        }

        // ---- f-MLP ----
        gemm1(sIn, sWgt + SEG_F1, bf1 + t * HID, sH1, l16, half, wv, lane);
        __syncthreads();
        gemm2(sH1, sWgt + SEG_F2, bf2 + t * HID, sH2, l16, half, wv, lane);
        __syncthreads();

        // jump = hf2 . C[t][r][d] + c0[t][r][d]
        float jump;
        {
            int cidx = (t * NUMR + r) * 21 + (d + GAP);
            const float* cp_ = Cjp + (size_t)cidx * HID;
            jump = c0jp[cidx];
            for (int k = 0; k < HID; ++k) jump += (float)sH2[tid * HID + k] * cp_[k];
        }

        // ---- per-path geometry / SDE update ----
        float nx = fmaxf(sqrtf(xt[0]*xt[0] + xt[1]*xt[1] + xt[2]*xt[2]), 1e-12f);
        float cz = fminf(1.f, fmaxf(-1.f, xt[2] / nx));
        float th = acosf(cz) - 0.5f * PI_F;
        float ph = atan2f(xt[1], xt[0]);
        float ct = cosf(th), st = sinf(th), cp = cosf(ph), sp = sinf(ph);
        float T00 = cp*ct, T01 = -sp, T02 = cp*st;
        float T10 = sp*ct, T11 =  cp, T12 = sp*st;
        float T20 = -st,              T22 = ct;

        float v10 = gu[0]*gtx[0] + gu[1]*gtx[3] + gu[2]*gtx[6];
        float v11 = gu[0]*gtx[1] + gu[1]*gtx[4] + gu[2]*gtx[7];
        float v12 = gu[0]*gtx[2] + gu[1]*gtx[5] + gu[2]*gtx[8];
        float v21 = v10*T01 + v11*T11;
        float v22 = v10*T02 + v11*T12 + v12*T22;
        float gux0 = -v22, gux1 = v21;                   // v @ G
        float guy0 = gu[3]*gty[0] + gu[4]*gty[3] + gu[5]*gty[6];
        float guy1 = gu[3]*gty[1] + gu[4]*gty[4] + gu[5]*gty[7];
        float guy2 = gu[3]*gty[2] + gu[4]*gty[5] + gu[5]*gty[8];

        float sdx = 1.0f / (float)(r + RMIN_I);
        float dbx0 = dBxt[(size_t)(t * BATCH + gid) * 2 + 0];
        float dbx1 = dBxt[(size_t)(t * BATCH + gid) * 2 + 1];
        float dby0 = dByt[(size_t)(t * BATCH + gid) * 3 + 0];
        float dby1 = dByt[(size_t)(t * BATCH + gid) * 3 + 1];
        float dby2 = dByt[(size_t)(t * BATCH + gid) * 3 + 2];

        float diff = sdx * (gux0*dbx0 + gux1*dbx1)
                   + (guy0*dby0 + guy1*dby1 + guy2*dby2)
                   + jump;
        u_pre += run * __expf(-fun) * diff;
        fun   += cfr[r] * DT_F;

        float dX0 = sdx * dbx0, dX1 = sdx * dbx1;
        float tht = dX0 + 0.5f * PI_F;
        float sT = sinf(tht), cT = cosf(tht), sP = sinf(dX1), cP = cosf(dX1);
        float c0_ = sT * cP - 1.f, c1_ = sT * sP, c2_ = cT;
        float dx30 = T00*c0_ + T01*c1_ + T02*c2_;
        float dx31 = T10*c0_ + T11*c1_ + T12*c2_;
        float dx32 = T20*c0_ +           T22*c2_;
        xt[0] += dx30; xt[1] += dx31; xt[2] += dx32;
        #pragma unroll
        for (int i = 0; i < 3; ++i)
            xin[i] += gtx[i*3+0]*dx30 + gtx[i*3+1]*dx31 + gtx[i*3+2]*dx32;
        #pragma unroll
        for (int i = 0; i < 3; ++i)
            yin[i] += gty[i*3+0]*dby0 + gty[i*3+1]*dby1 + gty[i*3+2]*dby2;

        reflect3(xin, gtx);
        reflect3(yin, gty);

        float e0 = xin[0]-yin[0], e1 = xin[1]-yin[1], e2 = xin[2]-yin[2];
        float dist = sqrtf(e0*e0 + e1*e1 + e2*e2);
        run = (dist < CAPEPS) ? 0.f : run;
        __syncthreads();   // sH2 / sWgt reads done before next step's TDM + tile writes
    }

    float e0 = xin[0]-yin[0], e1 = xin[1]-yin[1], e2 = xin[2]-yin[2];
    float u0v = __expf(-(e0*e0 + e1*e1 + e2*e2));
    out[gid]         = u_pre;
    out[BATCH + gid] = run * u0v * __expf(-fun);
}

// ---------------- host-side orchestration ----------------
extern "C" void kernel_launch(void* const* d_in, const int* in_sizes, int n_in,
                              void* d_out, int out_size, void* d_ws, size_t ws_size,
                              hipStream_t stream)
{
    (void)in_sizes; (void)n_in; (void)out_size; (void)ws_size;

    const float* u_in      = (const float*)d_in[0];
    const float* jump_r    = (const float*)d_in[1];
    const float* jump_l    = (const float*)d_in[2];
    const float* W1        = (const float*)d_in[3];
    const float* b1        = (const float*)d_in[4];
    const float* W2        = (const float*)d_in[5];
    const float* b2        = (const float*)d_in[6];
    const float* W3        = (const float*)d_in[7];
    const float* b3        = (const float*)d_in[8];
    const float* emb       = (const float*)d_in[9];
    const float* Wf1       = (const float*)d_in[10];
    const float* bf1       = (const float*)d_in[11];
    const float* Wf2       = (const float*)d_in[12];
    const float* bf2       = (const float*)d_in[13];
    const float* Wf3       = (const float*)d_in[14];
    const float* bf3       = (const float*)d_in[15];
    const int*   rt0       = (const int*)  d_in[16];
    const float* xt0       = (const float*)d_in[17];
    const float* yt0       = (const float*)d_in[18];
    const float* dBxt      = (const float*)d_in[19];
    const float* dByt      = (const float*)d_in[20];
    const float* jump_unif = (const float*)d_in[21];
    const float* size_unif = (const float*)d_in[22];
    const float* mc_unif   = (const float*)d_in[23];
    const float* jm        = (const float*)d_in[24];
    const float* cr        = (const float*)d_in[25];
    const float* cfr       = (const float*)d_in[26];

    char* ws = (char*)d_ws;
    size_t off = 0;
    auto alloc = [&](size_t bytes) -> void* {
        off = (off + 255) & ~(size_t)255;
        void* p = ws + off;
        off += bytes;
        return p;
    };
    signed char* ridx8 = (signed char*)alloc((size_t)NSTEP * BATCH);
    signed char* drt8  = (signed char*)alloc((size_t)NSTEP * BATCH);
    int*       hist    = (int*)      alloc((size_t)NUMR * 2 * GAP * 4);
    float*     mc_jump = (float*)    alloc((size_t)NUMR * PDIM * 4);
    float*     W3r     = (float*)    alloc((size_t)NSTEP * NUMR * HID * 6 * 4);
    float*     b3r     = (float*)    alloc((size_t)NSTEP * NUMR * 6 * 4);
    float*     Cjp     = (float*)    alloc((size_t)NSTEP * NUMR * 21 * HID * 4);
    float*     c0jp    = (float*)    alloc((size_t)NSTEP * NUMR * 21 * 4);
    _Float16*  Wpack   = (_Float16*) alloc((size_t)NSTEP * WBLK * 2);

    k_rchain<<<BATCH / 256, 256, 0, stream>>>(rt0, jump_unif, size_unif, jm, cr, ridx8, drt8);

    k_zero_i32<<<1, 256, 0, stream>>>(hist, NUMR * 2 * GAP);
    k_hist<<<(MC * NUMR + 255) / 256, 256, 0, stream>>>(mc_unif, jm, hist);
    k_mcbuild<<<NUMR, PDIM, 0, stream>>>(hist, jump_l, cr, mc_jump);

    k_w3r<<<NSTEP * NUMR, PDIM, 0, stream>>>(emb, W3, b3, W3r, b3r);
    k_wf3r<<<NSTEP * NUMR * 21, HID, 0, stream>>>(Wf3, bf3, jump_r, jump_l, mc_jump, Cjp, c0jp);
    k_pack<<<NSTEP, 256, 0, stream>>>(W1, W2, Wf1, Wf2, Wpack);

    k_main<<<BATCH / 128, 128, 0, stream>>>(ridx8, drt8, Wpack,
                                            b1, b2, bf1, bf2,
                                            W3r, b3r, Cjp, c0jp,
                                            u_in, xt0, yt0, dBxt, dByt, cfr,
                                            (float*)d_out);
}
